// S5Layer_28707561406817
// MI455X (gfx1250) — matrix operand order repaired and checked by reading, not compiled, not run on hardware
//
#include <hip/hip_runtime.h>

#define LL 16384
#define HH 1024
#define PP 1024
#define TT 128
#define NC (LL / TT)

typedef __attribute__((ext_vector_type(16))) __bf16 v16bf;
typedef __attribute__((ext_vector_type(8)))  float  v8f;

static __device__ __forceinline__ v8f wmma_bf16(v16bf a, v16bf b, v8f c) {
    return __builtin_amdgcn_wmma_f32_16x16x32_bf16(false, a, false, b, (short)0, c, false, false);
}

// Load 16 f32 K-values in this code's A order, same as B (K = k0+0..15 contiguous for this lane)
// and split each into bf16 hi/lo so that hi+lo ~= x to ~2^-16 relative.
static __device__ __forceinline__ void loadsplit_a(const float* __restrict__ rp,
                                                   v16bf& hi, v16bf& lo) {
    float4 a0 = *(const float4*)(rp);
    float4 a1 = *(const float4*)(rp + 4);
    float4 a2 = *(const float4*)(rp + 8);
    float4 a3 = *(const float4*)(rp + 12);
    float f[16] = {a0.x, a0.y, a0.z, a0.w, a1.x, a1.y, a1.z, a1.w,
                   a2.x, a2.y, a2.z, a2.w, a3.x, a3.y, a3.z, a3.w};
#pragma unroll
    for (int e = 0; e < 16; ++e) {
        float x = f[e];
        __bf16 h = (__bf16)x;
        hi[e] = h;
        lo[e] = (__bf16)(x - (float)h);
    }
}

// ---------------------------------------------------------------------------
// 1) Discretization + weight conversion to split-bf16 planes.
//    B-plane layout (P,H): row p contiguous over h  -> WMMA B operand for GEMM1.
//    C-plane layout (H,P): row h contiguous over p  -> WMMA B operand for GEMM2.
//    Factor 2 (conj_sym) folded into C_re; minus sign folded into C_im.
// ---------------------------------------------------------------------------
__global__ void s5_prep(const float* __restrict__ Bc, const float* __restrict__ Cc,
                        const float* __restrict__ LamR, const float* __restrict__ LamI,
                        const float* __restrict__ logstep,
                        float* __restrict__ lamRe, float* __restrict__ lamIm,
                        __bf16* __restrict__ BreHi, __bf16* __restrict__ BreLo,
                        __bf16* __restrict__ BimHi, __bf16* __restrict__ BimLo,
                        __bf16* __restrict__ CreHi, __bf16* __restrict__ CreLo,
                        __bf16* __restrict__ CimHi, __bf16* __restrict__ CimLo) {
    int gid = blockIdx.x * blockDim.x + threadIdx.x;
    if (gid >= PP * HH) return;
    int p = gid / HH;

    float lr = fminf(LamR[p], -1e-4f);
    float li = LamI[p];
    float s  = expf(logstep[p]);
    float er = expf(lr * s);
    float sb, cb;
    sincosf(li * s, &sb, &cb);
    float lbr = er * cb, lbi = er * sb;          // Lambda_bar
    if ((gid % HH) == 0) { lamRe[p] = lbr; lamIm[p] = lbi; }

    // f = (Lambda_bar - 1) / Lambda
    float nr = lbr - 1.0f, ni = lbi;
    float invd = 1.0f / (lr * lr + li * li);
    float fr = (nr * lr + ni * li) * invd;
    float fi = (ni * lr - nr * li) * invd;

    float br = Bc[2 * gid + 0], bi = Bc[2 * gid + 1];
    float wr = fr * br - fi * bi;
    float wi = fr * bi + fi * br;
    { __bf16 h = (__bf16)wr; BreHi[gid] = h; BreLo[gid] = (__bf16)(wr - (float)h); }
    { __bf16 h = (__bf16)wi; BimHi[gid] = h; BimLo[gid] = (__bf16)(wi - (float)h); }

    // gid also enumerates (h,p) for C (H*P == P*H)
    float cr =  2.0f * Cc[2 * gid + 0];
    float ci = -2.0f * Cc[2 * gid + 1];
    { __bf16 h = (__bf16)cr; CreHi[gid] = h; CreLo[gid] = (__bf16)(cr - (float)h); }
    { __bf16 h = (__bf16)ci; CimHi[gid] = h; CimLo[gid] = (__bf16)(ci - (float)h); }
}

// ---------------------------------------------------------------------------
// 2) Bu = u @ B_bar^T  (re and im planes).  One wave per 32x32 output tile.
//    N-subtiles processed serially inside the k-loop to keep B fragments'
//    live ranges short (avoid scratch spills).
// ---------------------------------------------------------------------------
__global__ void __launch_bounds__(32)
s5_gemm_in(const float* __restrict__ u,
           const __bf16* __restrict__ BreHi, const __bf16* __restrict__ BreLo,
           const __bf16* __restrict__ BimHi, const __bf16* __restrict__ BimLo,
           float* __restrict__ BuRe, float* __restrict__ BuIm) {
    const int lane = threadIdx.x;
    const int n0 = blockIdx.x * 32;     // over P
    const int m0 = blockIdx.y * 32;     // over L
    const int ln = lane & 15;
    const int hhf = (lane >> 4) & 1;

    v8f z;
#pragma unroll
    for (int e = 0; e < 8; ++e) z[e] = 0.0f;
    v8f aR[2][2], aI[2][2];
#pragma unroll
    for (int mi = 0; mi < 2; ++mi)
#pragma unroll
        for (int ni = 0; ni < 2; ++ni) { aR[mi][ni] = z; aI[mi][ni] = z; }

    const int ka = hhf * 16;            // A: contiguous 16-K run per half-wave
    const int kb = hhf * 16;            // B: contiguous 16-K run per half-wave

    for (int kk = 0; kk < HH; kk += 32) {
        v16bf ah[2], al[2];
#pragma unroll
        for (int mi = 0; mi < 2; ++mi)
            loadsplit_a(u + (size_t)(m0 + mi * 16 + ln) * HH + kk + ka, ah[mi], al[mi]);

#pragma unroll
        for (int ni = 0; ni < 2; ++ni) {
            size_t off = (size_t)(n0 + ni * 16 + ln) * HH + kk + kb;
            v16bf rh = *(const v16bf*)(BreHi + off);
            v16bf rl = *(const v16bf*)(BreLo + off);
            v16bf ih = *(const v16bf*)(BimHi + off);
            v16bf il = *(const v16bf*)(BimLo + off);
#pragma unroll
            for (int mi = 0; mi < 2; ++mi) {
                v8f r = aR[mi][ni];
                r = wmma_bf16(ah[mi], rh, r);
                r = wmma_bf16(ah[mi], rl, r);
                r = wmma_bf16(al[mi], rh, r);
                aR[mi][ni] = r;
                v8f q = aI[mi][ni];
                q = wmma_bf16(ah[mi], ih, q);
                q = wmma_bf16(ah[mi], il, q);
                q = wmma_bf16(al[mi], ih, q);
                aI[mi][ni] = q;
            }
        }
    }
#pragma unroll
    for (int mi = 0; mi < 2; ++mi)
#pragma unroll
        for (int ni = 0; ni < 2; ++ni) {
            int col = n0 + ni * 16 + ln;
#pragma unroll
            for (int r = 0; r < 8; ++r) {
                int row = m0 + mi * 16 + r + hhf * 8;
                size_t o = (size_t)row * PP + col;
                BuRe[o] = aR[mi][ni][r];
                BuIm[o] = aI[mi][ni][r];
            }
        }
}

// ---------------------------------------------------------------------------
// 3) Chunked diagonal complex scan: x_t = lam*x_{t-1} + Bu_t   (in place)
// ---------------------------------------------------------------------------
__global__ void s5_scan_local(float* __restrict__ BuRe, float* __restrict__ BuIm,
                              const float* __restrict__ lamRe, const float* __restrict__ lamIm,
                              float* __restrict__ lastRe, float* __restrict__ lastIm) {
    int gid = blockIdx.x * blockDim.x + threadIdx.x;
    if (gid >= NC * PP) return;
    int c = gid / PP, p = gid % PP;
    float lr = lamRe[p], li = lamIm[p];
    float xr = 0.0f, xi = 0.0f;
    size_t base = (size_t)c * TT * PP + p;
    for (int i = 0; i < TT; ++i) {
        size_t idx = base + (size_t)i * PP;
        float br = BuRe[idx], bi = BuIm[idx];
        float nxr = lr * xr - li * xi + br;
        float nxi = lr * xi + li * xr + bi;
        xr = nxr; xi = nxi;
        BuRe[idx] = xr; BuIm[idx] = xi;
    }
    lastRe[c * PP + p] = xr;
    lastIm[c * PP + p] = xi;
}

__global__ void s5_scan_carry(const float* __restrict__ lamRe, const float* __restrict__ lamIm,
                              const float* __restrict__ lastRe, const float* __restrict__ lastIm,
                              float* __restrict__ cinRe, float* __restrict__ cinIm) {
    int p = blockIdx.x * blockDim.x + threadIdx.x;
    if (p >= PP) return;
    float wr = lamRe[p], wi = lamIm[p];
#pragma unroll
    for (int k = 0; k < 7; ++k) {               // lam^(2^7) = lam^TT
        float nr = wr * wr - wi * wi;
        float ni = 2.0f * wr * wi;
        wr = nr; wi = ni;
    }
    float xr = 0.0f, xi = 0.0f;
    for (int c = 0; c < NC; ++c) {
        cinRe[c * PP + p] = xr;
        cinIm[c * PP + p] = xi;
        float br = lastRe[c * PP + p], bi = lastIm[c * PP + p];
        float nxr = wr * xr - wi * xi + br;
        float nxi = wr * xi + wi * xr + bi;
        xr = nxr; xi = nxi;
    }
}

__global__ void s5_scan_apply(float* __restrict__ BuRe, float* __restrict__ BuIm,
                              const float* __restrict__ lamRe, const float* __restrict__ lamIm,
                              const float* __restrict__ cinRe, const float* __restrict__ cinIm) {
    int gid = blockIdx.x * blockDim.x + threadIdx.x;
    if (gid >= NC * PP) return;
    int c = gid / PP, p = gid % PP;
    float ar = cinRe[c * PP + p], ai = cinIm[c * PP + p];
    if (ar == 0.0f && ai == 0.0f) return;       // chunk 0 (and any dead carries)
    float lr = lamRe[p], li = lamIm[p];
    size_t base = (size_t)c * TT * PP + p;
    for (int i = 0; i < TT; ++i) {
        float nar = lr * ar - li * ai;          // lam^(i+1) * carry_in
        float nai = lr * ai + li * ar;
        ar = nar; ai = nai;
        size_t idx = base + (size_t)i * PP;
        BuRe[idx] += ar;
        BuIm[idx] += ai;
    }
}

// ---------------------------------------------------------------------------
// 4) out = 2*Re(x @ C~^T) + u*D  (2 and sign pre-folded into C planes)
// ---------------------------------------------------------------------------
__global__ void __launch_bounds__(32)
s5_gemm_out(const float* __restrict__ XRe, const float* __restrict__ XIm,
            const __bf16* __restrict__ CreHi, const __bf16* __restrict__ CreLo,
            const __bf16* __restrict__ CimHi, const __bf16* __restrict__ CimLo,
            const float* __restrict__ u, const float* __restrict__ Dv,
            float* __restrict__ out) {
    const int lane = threadIdx.x;
    const int n0 = blockIdx.x * 32;     // over H
    const int m0 = blockIdx.y * 32;     // over L
    const int ln = lane & 15;
    const int hhf = (lane >> 4) & 1;

    v8f z;
#pragma unroll
    for (int e = 0; e < 8; ++e) z[e] = 0.0f;
    v8f acc[2][2];
#pragma unroll
    for (int mi = 0; mi < 2; ++mi)
#pragma unroll
        for (int ni = 0; ni < 2; ++ni) acc[mi][ni] = z;

    const int ka = hhf * 16;
    const int kb = hhf * 16;

    for (int kk = 0; kk < PP; kk += 32) {
        v16bf arh[2], arl[2], aih[2], ail[2];
#pragma unroll
        for (int mi = 0; mi < 2; ++mi) {
            size_t rbase = (size_t)(m0 + mi * 16 + ln) * PP + kk + ka;
            loadsplit_a(XRe + rbase, arh[mi], arl[mi]);
            loadsplit_a(XIm + rbase, aih[mi], ail[mi]);
        }
#pragma unroll
        for (int ni = 0; ni < 2; ++ni) {
            size_t off = (size_t)(n0 + ni * 16 + ln) * PP + kk + kb;
            v16bf crh = *(const v16bf*)(CreHi + off);
            v16bf crl = *(const v16bf*)(CreLo + off);
            v16bf cih = *(const v16bf*)(CimHi + off);
            v16bf cil = *(const v16bf*)(CimLo + off);
#pragma unroll
            for (int mi = 0; mi < 2; ++mi) {
                v8f a = acc[mi][ni];
                a = wmma_bf16(arh[mi], crh, a);
                a = wmma_bf16(arh[mi], crl, a);
                a = wmma_bf16(arl[mi], crh, a);
                a = wmma_bf16(aih[mi], cih, a);
                a = wmma_bf16(aih[mi], cil, a);
                a = wmma_bf16(ail[mi], cih, a);
                acc[mi][ni] = a;
            }
        }
    }
#pragma unroll
    for (int mi = 0; mi < 2; ++mi)
#pragma unroll
        for (int ni = 0; ni < 2; ++ni) {
            int col = n0 + ni * 16 + ln;
            float dv = Dv[col];
#pragma unroll
            for (int r = 0; r < 8; ++r) {
                int row = m0 + mi * 16 + r + hhf * 8;
                size_t o = (size_t)row * HH + col;
                out[o] = acc[mi][ni][r] + u[o] * dv;
            }
        }
}

extern "C" void kernel_launch(void* const* d_in, const int* in_sizes, int n_in,
                              void* d_out, int out_size, void* d_ws, size_t ws_size,
                              hipStream_t stream) {
    (void)in_sizes; (void)n_in; (void)out_size; (void)ws_size;
    const float* u    = (const float*)d_in[0];
    const float* LamR = (const float*)d_in[1];
    const float* LamI = (const float*)d_in[2];
    const float* Bc   = (const float*)d_in[3];
    const float* Cc   = (const float*)d_in[4];
    const float* Dv   = (const float*)d_in[5];
    const float* lgst = (const float*)d_in[6];
    float* out = (float*)d_out;

    // workspace carve-up (~153 MB)
    float* BuRe   = (float*)d_ws;
    float* BuIm   = BuRe   + (size_t)LL * PP;
    float* lastRe = BuIm   + (size_t)LL * PP;
    float* lastIm = lastRe + (size_t)NC * PP;
    float* cinRe  = lastIm + (size_t)NC * PP;
    float* cinIm  = cinRe  + (size_t)NC * PP;
    float* lamRe  = cinIm  + (size_t)NC * PP;
    float* lamIm  = lamRe  + PP;
    __bf16* BreHi = (__bf16*)(lamIm + PP);
    __bf16* BreLo = BreHi + (size_t)PP * HH;
    __bf16* BimHi = BreLo + (size_t)PP * HH;
    __bf16* BimLo = BimHi + (size_t)PP * HH;
    __bf16* CreHi = BimLo + (size_t)PP * HH;
    __bf16* CreLo = CreHi + (size_t)PP * HH;
    __bf16* CimHi = CreLo + (size_t)PP * HH;
    __bf16* CimLo = CimHi + (size_t)PP * HH;

    s5_prep<<<(PP * HH) / 256, 256, 0, stream>>>(
        Bc, Cc, LamR, LamI, lgst, lamRe, lamIm,
        BreHi, BreLo, BimHi, BimLo, CreHi, CreLo, CimHi, CimLo);

    s5_gemm_in<<<dim3(PP / 32, LL / 32), 32, 0, stream>>>(
        u, BreHi, BreLo, BimHi, BimLo, BuRe, BuIm);

    s5_scan_local<<<(NC * PP) / 256, 256, 0, stream>>>(
        BuRe, BuIm, lamRe, lamIm, lastRe, lastIm);

    s5_scan_carry<<<PP / 256, 256, 0, stream>>>(
        lamRe, lamIm, lastRe, lastIm, cinRe, cinIm);

    s5_scan_apply<<<(NC * PP) / 256, 256, 0, stream>>>(
        BuRe, BuIm, lamRe, lamIm, cinRe, cinIm);

    s5_gemm_out<<<dim3(HH / 32, LL / 32), 32, 0, stream>>>(
        BuRe, BuIm, CreHi, CreLo, CimHi, CimLo, u, Dv, out);
}
